// VarianceAdaptor_17145509445864
// MI455X (gfx1250) — matrix operand order repaired and checked by reading, not compiled, not run on hardware
//
#include <hip/hip_runtime.h>
#include <hip/hip_bf16.h>
#include <cmath>

// ---------------- problem constants ----------------
constexpr int B  = 32;
constexpr int T  = 512;
constexpr int H  = 256;
constexpr int L  = 2048;
constexpr int NB = 256;
constexpr int NL = 4;

// d_out offsets (float elements)
constexpr long OFF_OUT  = 0;
constexpr long OFF_DUR  = (long)B * L * H;            // 16,777,216
constexpr long OFF_PIT  = OFF_DUR + (long)B * T;      // +16384
constexpr long OFF_EN   = OFF_PIT + (long)B * L;      // +65536
constexpr long OFF_MLEN = OFF_EN  + (long)B * L;      // +65536
constexpr long OFF_MASK = OFF_MLEN + B;               // +32

// workspace offsets (bytes)
constexpr size_t WS_XBF  = 0;                                  // B*T*H bf16 = 8 MB
constexpr size_t WS_WT   = WS_XBF + (size_t)B * T * H * 2;     // 3*NL*H*H bf16 = 1.5 MB
constexpr size_t WS_IDX  = WS_WT  + (size_t)3 * NL * H * H * 2;// B*L int = 256 KB
constexpr size_t WS_MLEN = WS_IDX + (size_t)B * L * 4;         // B int

typedef __attribute__((ext_vector_type(16))) __bf16 v16bf;
typedef __attribute__((ext_vector_type(8)))  __bf16 v8bf;
typedef __attribute__((ext_vector_type(8)))  float  v8f;
typedef __attribute__((ext_vector_type(4)))  unsigned int u32x4;
typedef __attribute__((ext_vector_type(8)))  int i32x8;
typedef __attribute__((ext_vector_type(4)))  int i32x4;

// ---------------- prep kernels ----------------

// Wt[s][l][out][in] (bf16) = W_s[l][in][out] (f32)
__global__ __launch_bounds__(256) void prep_weights_kernel(
    const float* __restrict__ W0, const float* __restrict__ W1,
    const float* __restrict__ W2, __bf16* __restrict__ wt)
{
    int i = blockIdx.x * 256 + threadIdx.x;          // < 3*NL*H*H
    int s  = i / (NL * H * H);
    int r  = i - s * (NL * H * H);
    int l  = r / (H * H);
    int r2 = r - l * (H * H);
    int o  = r2 / H;
    int in = r2 - o * H;
    const float* W = (s == 0) ? W0 : ((s == 1) ? W1 : W2);
    wt[i] = (__bf16)W[l * H * H + in * H + o];
}

__global__ __launch_bounds__(256) void prep_x_kernel(
    const float* __restrict__ x, __bf16* __restrict__ xbf, int n)
{
    int i = blockIdx.x * 256 + threadIdx.x;
    if (i < n) xbf[i] = (__bf16)x[i];
}

// ---------------- length regulator ----------------
__global__ __launch_bounds__(256) void regulate_kernel(
    const int* __restrict__ dur, const int* __restrict__ maxlenp,
    int* __restrict__ idxarr, int* __restrict__ mlen_ws,
    float* __restrict__ outbase)
{
    __shared__ int cum[T];
    int b   = blockIdx.x;
    int tid = threadIdx.x;
    for (int t = tid; t < T; t += blockDim.x) cum[t] = dur[b * T + t];
    __syncthreads();
    if (tid == 0) {
        int s = 0;
        for (int t = 0; t < T; ++t) { s += cum[t]; cum[t] = s; }
    }
    __syncthreads();
    int cap = maxlenp[0];                 // == L
    if (cap > L) cap = L;
    int ml = cum[T - 1] < cap ? cum[T - 1] : cap;
    if (tid == 0) {
        mlen_ws[b] = ml;
        outbase[OFF_MLEN + b] = (float)ml;
    }
    for (int m = tid; m < L; m += blockDim.x) {
        // searchsorted(cum, m, side='right'): first i with cum[i] > m
        int lo = 0, hi = T;
        while (lo < hi) {
            int mid = (lo + hi) >> 1;
            if (cum[mid] <= m) lo = mid + 1; else hi = mid;
        }
        int idx = lo < (T - 1) ? lo : (T - 1);
        idxarr[b * L + m] = idx;
        outbase[OFF_MASK + (long)b * L + m] = (m >= ml) ? 1.0f : 0.0f;
    }
}

// ---------------- out = xe + p_emb + e_emb ----------------
__global__ __launch_bounds__(256) void assemble_out_kernel(
    const float* __restrict__ x, const int* __restrict__ idxarr,
    const int* __restrict__ mlen, const float* __restrict__ pt,
    const float* __restrict__ et, const float* __restrict__ pemb,
    const float* __restrict__ eemb, float* __restrict__ out)
{
    int i  = blockIdx.x * 256 + threadIdx.x;   // over B*L*(H/4)
    int bm = i >> 6;
    int h  = (i & 63) << 2;
    int b  = bm / L;
    int m  = bm - b * L;
    bool masked = (m >= mlen[b]);
    int src = idxarr[bm];
    const float4 xv = *(const float4*)&x[((long)b * T + src) * H + h];
    int pi = (int)ceilf(pt[bm] * (float)NB);
    int ei = (int)ceilf(et[bm] * (float)NB);
    const float4 pv = *(const float4*)&pemb[(long)pi * H + h];
    const float4 ev = *(const float4*)&eemb[(long)ei * H + h];
    float4 o;
    o.x = (masked ? 0.f : xv.x) + pv.x + ev.x;
    o.y = (masked ? 0.f : xv.y) + pv.y + ev.y;
    o.z = (masked ? 0.f : xv.z) + pv.z + ev.z;
    o.w = (masked ? 0.f : xv.w) + pv.w + ev.w;
    *(float4*)&out[(long)bm * H + h] = o;
}

// ---------------- TDM: stage one H x H bf16 weight matrix into LDS ----------
// D# per CDNA5 ISA ch.8: 2-D tensor, data_size=2B, tensor_dim0/1 = 256,
// tile_dim0/1 = 256, tensor_dim0_stride = 256 -> linear 128KB copy.
__device__ inline void tdm_load_weights(const __bf16* gsrc, unsigned ldsoff)
{
    unsigned long long ga = (unsigned long long)(uintptr_t)gsrc;
    u32x4 g0 = {};
    g0[0] = 1u;                                   // count=1, user mode
    g0[1] = ldsoff;                               // lds_addr
    g0[2] = (unsigned)ga;                         // global_addr[31:0]
    g0[3] = (unsigned)((ga >> 32) & 0x01FFFFFFu)  // global_addr[56:32]
            | (2u << 30);                         // type = 2 ("image")
    i32x8 g1 = {};
    g1[0] = 0x00010000;            // workgroup_mask=0, data_size=1 (2 bytes)
    g1[1] = (int)(256u << 16);     // tensor_dim0[15:0] = 256
    g1[2] = (int)(256u << 16);     // tensor_dim0[31:16]=0, tensor_dim1[15:0]=256
    g1[3] = (int)(256u << 16);     // tensor_dim1[31:16]=0, tile_dim0 = 256
    g1[4] = 256;                   // tile_dim1 = 256, tile_dim2 = 0
    g1[5] = 256;                   // tensor_dim0_stride = 256
    i32x4 g2 = {};
    i32x4 g3 = {};
#if defined(__clang_major__) && (__clang_major__ >= 23)
    i32x8 g4 = {};
    __builtin_amdgcn_tensor_load_to_lds(g0, g1, g2, g3, g4, 0);
#else
    __builtin_amdgcn_tensor_load_to_lds(g0, g1, g2, g3, 0);
#endif
}

// ---------------- WMMA predictor ----------------
// One wave computes a 16-row tile through 4 layers of H x H (bf16 WMMA,
// f32 accum). Layer weights are DMA'd (TDM) into a shared LDS buffer once
// per block per layer; activations ping-pong through per-wave LDS tiles.
#define NWAVE 8
constexpr size_t ACT_BYTES  = (size_t)NWAVE * 2 * 16 * H * 2;  // 128 KB
constexpr size_t WTB_BYTES  = (size_t)H * H * 2;               // 128 KB
constexpr size_t SMEM_BYTES = ACT_BYTES + WTB_BYTES;           // 256 KB

__device__ inline void layer_mm(const __bf16* __restrict__ arow, bool zerorow,
                                const __bf16* __restrict__ wtl,
                                const float* __restrict__ bl,
                                __bf16* __restrict__ wdst, int lane)
{
    const int hiHalf   = lane >> 4;        // 0 for lanes 0-15, 1 for 16-31
    const int colq     = lane & 15;
    const int baseAoff = hiHalf * 16;      // contiguous K split: lane half h holds k = 16h..16h+15
    const int baseBoff = hiHalf * 16;      // same contiguous K split for B (k = 16h..16h+15)
    const v16bf vzero = {};

    for (int nt = 0; nt < 16; ++nt) {
        v8f acc = {};
        const __bf16* bcol = wtl + (nt * 16 + colq) * H + baseBoff;
#pragma unroll
        for (int kt = 0; kt < 8; ++kt) {
            const int k0 = kt * 32;
            // A fragment: two contiguous 16B chunks per lane
            v8bf alo = *(const v8bf*)(arow + k0 + baseAoff);
            v8bf ahi = *(const v8bf*)(arow + k0 + baseAoff + 8);
            v16bf a;
#pragma unroll
            for (int e = 0; e < 8; ++e) { a[e] = alo[e]; a[e + 8] = ahi[e]; }
            if (zerorow) a = vzero;
            // B fragment: one contiguous 32B chunk per lane from LDS Wt[out][in]
            v16bf bf = *(const v16bf*)(bcol + k0);
            acc = __builtin_amdgcn_wmma_f32_16x16x32_bf16(
                false, a, false, bf, (short)0, acc, false, false);
        }
        const float bb = bl[nt * 16 + colq];
#pragma unroll
        for (int r = 0; r < 8; ++r) {
            float f = acc[r] + bb;
            f = fmaxf(f, 0.0f);                       // ReLU every layer
            wdst[(r + hiHalf * 8) * H + (nt * 16 + colq)] = (__bf16)f;
        }
    }
}

__global__ __launch_bounds__(NWAVE * 32) void predictor_kernel(
    const __bf16* __restrict__ xbf,     // [B*T, H] bf16 activations
    const __bf16* __restrict__ wt,      // [NL][H(out)][H(in)] bf16
    const float*  __restrict__ bias,    // [NL][H]
    const float*  __restrict__ wvec,    // [H]
    const float*  __restrict__ b2,      // [1]
    const unsigned char* __restrict__ srcmask, // [B*T]   (mode 0)
    const int*    __restrict__ idxarr,  // [B*L]          (mode 1/2)
    const int*    __restrict__ mlen,    // [B]            (mode 1/2)
    const float*  __restrict__ srcseq,  // [B*T*3]        (mode 0)
    float* __restrict__ outv,           // [M]
    int mode)
{
    extern __shared__ char smem[];
    __bf16* sact = (__bf16*)smem;                 // [NWAVE][2][16*H]
    __bf16* swt  = (__bf16*)(smem + ACT_BYTES);   // [H][H] current layer W^T

    const int wid  = threadIdx.x >> 5;
    const int lane = threadIdx.x & 31;
    const int rowt = lane & 15;
    const int row_base = (blockIdx.x * NWAVE + wid) * 16;
    const int g = row_base + rowt;

    // layer-0 A source (length-regulator gather folded in for modes 1/2)
    const __bf16* arow;
    bool zerorow = false;
    if (mode == 0) {
        arow = xbf + (long)g * H;
    } else {
        int b = g / L;
        int m = g - b * L;
        int src = idxarr[g];
        zerorow = (m >= mlen[b]);
        arow = xbf + ((long)b * T + src) * H;
    }

    __bf16* buf0 = sact + (wid * 2 + 0) * (16 * H);
    __bf16* buf1 = sact + (wid * 2 + 1) * (16 * H);
    const unsigned swt_off = (unsigned)(uintptr_t)swt;

    for (int l = 0; l < NL; ++l) {
        __syncthreads();                    // previous layer done reading swt
        if (wid == 0) {
            tdm_load_weights(wt + l * H * H, swt_off);
            __builtin_amdgcn_s_wait_tensorcnt(0);
        }
        __syncthreads();                    // layer weights visible in LDS
        __bf16* wdst = (l & 1) ? buf1 : buf0;
        if (l == 0) {
            layer_mm(arow, zerorow, swt, bias, wdst, lane);
        } else {
            const __bf16* asrc = ((l & 1) ? buf0 : buf1) + rowt * H;
            layer_mm(asrc, false, swt, bias + l * H, wdst, lane);
        }
    }

    // head: out[m] = dot(act[m], w) + b2, two lanes per row
    const int r = lane >> 1;
    const int p = lane & 1;
    const __bf16* fin = buf1 + r * H + p * 128;   // (NL-1)&1 == 1
    const float* wv = wvec + p * 128;
    float s = 0.f;
#pragma unroll 4
    for (int h = 0; h < 128; ++h) s += (float)fin[h] * wv[h];
    s += __shfl_xor(s, 1, 32);
    if (p == 0) {
        int gg = row_base + r;
        float val = s + b2[0];
        if (mode == 0) {
            if (srcmask[gg]) val = 0.f;
            outv[gg] = (tanhf(val) + 1.0f) * srcseq[gg * 3 + 2];
        } else {
            int b = gg / L;
            int m = gg - b * L;
            if (m >= mlen[b]) val = 0.f;
            if (mode == 1) val = fmaxf(val, 0.f);
            outv[gg] = val;
        }
    }
}

// ---------------- host launcher ----------------
extern "C" void kernel_launch(void* const* d_in, const int* in_sizes, int n_in,
                              void* d_out, int out_size, void* d_ws, size_t ws_size,
                              hipStream_t stream)
{
    const float* x        = (const float*)d_in[0];
    const float* src_seq  = (const float*)d_in[1];
    const int*   dur_t    = (const int*)d_in[2];
    const float* pitch_t  = (const float*)d_in[3];
    const float* energy_t = (const float*)d_in[4];
    const unsigned char* src_mask = (const unsigned char*)d_in[5];
    const int*   max_len  = (const int*)d_in[6];
    const float* dur_W  = (const float*)d_in[7];
    const float* dur_b  = (const float*)d_in[8];
    const float* dur_w  = (const float*)d_in[9];
    const float* dur_b2 = (const float*)d_in[10];
    const float* pit_W  = (const float*)d_in[11];
    const float* pit_b  = (const float*)d_in[12];
    const float* pit_w  = (const float*)d_in[13];
    const float* pit_b2 = (const float*)d_in[14];
    const float* en_W   = (const float*)d_in[15];
    const float* en_b   = (const float*)d_in[16];
    const float* en_w   = (const float*)d_in[17];
    const float* en_b2  = (const float*)d_in[18];
    const float* p_emb  = (const float*)d_in[19];
    const float* e_emb  = (const float*)d_in[20];

    float* out = (float*)d_out;
    char* ws = (char*)d_ws;
    __bf16* xbf    = (__bf16*)(ws + WS_XBF);
    __bf16* wt     = (__bf16*)(ws + WS_WT);
    int*    idxarr = (int*)(ws + WS_IDX);
    int*    mlenws = (int*)(ws + WS_MLEN);

    // allow 256 KB dynamic LDS for the predictor (CDNA5: 320 KB per WGP)
    (void)hipFuncSetAttribute((const void*)predictor_kernel,
                              hipFuncAttributeMaxDynamicSharedMemorySize,
                              (int)SMEM_BYTES);

    // 1) weights -> bf16 transposed
    prep_weights_kernel<<<(3 * NL * H * H) / 256, 256, 0, stream>>>(
        dur_W, pit_W, en_W, wt);
    // 2) x -> bf16
    prep_x_kernel<<<(B * T * H) / 256, 256, 0, stream>>>(x, xbf, B * T * H);
    // 3) length regulator: idx, mel_len, mel_mask
    regulate_kernel<<<B, 256, 0, stream>>>(dur_t, max_len, idxarr, mlenws, out);
    // 4) out = xe + p_emb + e_emb
    assemble_out_kernel<<<(B * L * (H / 4)) / 256, 256, 0, stream>>>(
        x, idxarr, mlenws, pitch_t, energy_t, p_emb, e_emb, out + OFF_OUT);
    // 5) duration predictor (rows = B*T)
    predictor_kernel<<<(B * T) / (16 * NWAVE), NWAVE * 32, SMEM_BYTES, stream>>>(
        xbf, wt, dur_b, dur_w, dur_b2, src_mask, nullptr, nullptr,
        src_seq, out + OFF_DUR, 0);
    // 6) pitch predictor (rows = B*L, gathered)
    predictor_kernel<<<(B * L) / (16 * NWAVE), NWAVE * 32, SMEM_BYTES, stream>>>(
        xbf, wt + NL * H * H, pit_b, pit_w, pit_b2, nullptr, idxarr, mlenws,
        nullptr, out + OFF_PIT, 1);
    // 7) energy predictor (rows = B*L, gathered)
    predictor_kernel<<<(B * L) / (16 * NWAVE), NWAVE * 32, SMEM_BYTES, stream>>>(
        xbf, wt + 2 * NL * H * H, en_b, en_w, en_b2, nullptr, idxarr, mlenws,
        nullptr, out + OFF_EN, 2);
}
